// RWKVModel_37864431681711
// MI455X (gfx1250) — matrix operand order repaired and checked by reading, not compiled, not run on hardware
//
#include <hip/hip_runtime.h>
#include <hip/hip_bf16.h>

// RWKV-ish model, B=4 T=4096 H=2048 D=4.
// Compute-bound (~550 GFLOP vs ~0.5GB traffic) -> everything on v_wmma_f32_16x16x32_bf16
// with bf16x3 split (hi/lo) for ~fp32 accuracy.

#define HH 2048
#define TT 4096
#define BBATCH 4
#define DDEP 4
#define BM 128
#define BN 64
#define KC 32
#define LDK 40  // KC + 8 pad elems; row stride = 80 bytes (16B aligned, good bank spread)

typedef __attribute__((ext_vector_type(16))) __bf16 bf16x16;
typedef __attribute__((ext_vector_type(8)))  float  f32x8;

static_assert((BBATCH * TT) % BM == 0, "");
static_assert(HH % BN == 0 && HH % KC == 0, "");

__device__ __forceinline__ f32x8 wmma_bf16(bf16x16 a, bf16x16 b, f32x8 c) {
  return __builtin_amdgcn_wmma_f32_16x16x32_bf16(false, a, false, b, (short)0, c,
                                                 false, false);
}

// A fragment (16x32, 16-bit): lane l: row m=l&15, half=l>>4, two 16B chunks at
// K = 8*half and K = 16 + 8*half   (per ISA 7.12.2 table).
__device__ __forceinline__ bf16x16 load_frag_a(const __bf16* row, int hf) {
  union { bf16x16 v; uint4 q[2]; } f;
  f.q[0] = *(const uint4*)(row + 8 * hf);
  f.q[1] = *(const uint4*)(row + 16 + 8 * hf);
  return f.v;
}

// B fragment (32x16, 16-bit): lane l: col n=l&15, half=l>>4, two 16B chunks at
// K = 8*half and K = 16 + 8*half (same K order as A).
__device__ __forceinline__ bf16x16 load_frag_b(const __bf16* row, int hf) {
  union { bf16x16 v; uint4 q[2]; } f;
  f.q[0] = *(const uint4*)(row + 8 * hf);
  f.q[1] = *(const uint4*)(row + 16 + 8 * hf);
  return f.v;
}

union PK4 { __bf16 b[4]; uint2 u; };

// D = A @ W.T for NW weight matrices sharing the A tile.
// EPI=true: A=x, weights {Wv,Wk,Wr}; epilogue computes a = sigmoid(r)*(sum_d lw_d*decayed_d + k*v*sum(lw))
//           and stashes kv at t=T-1 into kv_last. EPI=false: plain store of product 0.
template <int NW, bool EPI>
__launch_bounds__(256) __global__
void gemm_kernel(const float* __restrict__ A, const float* __restrict__ W0,
                 const float* __restrict__ W1, const float* __restrict__ W2,
                 const float* __restrict__ lw,       // [M][4]   (EPI)
                 const float* __restrict__ decayed,  // [B][D][H](EPI)
                 float* __restrict__ out0,           // EPI: a_buf, else: out
                 float* __restrict__ kv_last)        // EPI: [B][H]
{
  __shared__ __bf16 Ahi[2][BM][LDK];
  __shared__ __bf16 Alo[2][BM][LDK];
  __shared__ __bf16 Bhi[2][NW][BN][LDK];
  __shared__ __bf16 Blo[2][NW][BN][LDK];
  __shared__ float  dec_s[DDEP][BN];

  const int tid  = threadIdx.x;
  const int lane = tid & 31;
  const int wave = tid >> 5;
  const int waveM = wave & 3;   // 4 wave-rows of 32
  const int waveN = wave >> 2;  // 2 wave-cols of 32
  const int hf  = lane >> 4;
  const int l16 = lane & 15;

  const int row0 = blockIdx.x * BM;
  const int col0 = blockIdx.y * BN;

  const float* const Wp[3] = {W0, W1, W2};

  if constexpr (EPI) {
    // one batch per block tile (BM divides T), preload decayed slice [4][BN]
    const int b = row0 / TT;
    const int d = tid / BN, c = tid % BN;  // 256 == 4*64 exactly
    dec_s[d][c] = decayed[(size_t)(b * DDEP + d) * HH + col0 + c];
  }

  const f32x8 zero = {0.f, 0.f, 0.f, 0.f, 0.f, 0.f, 0.f, 0.f};
  f32x8 acc[NW][2][2];
#pragma unroll
  for (int w = 0; w < NW; ++w)
#pragma unroll
    for (int mf = 0; mf < 2; ++mf)
#pragma unroll
      for (int nf = 0; nf < 2; ++nf) acc[w][mf][nf] = zero;

  float4 ga[4];
  float4 gb[NW][2];

  auto load_global = [&](int kc) {
    const int k0 = kc * KC;
#pragma unroll
    for (int i = 0; i < 4; ++i) {
      int s = tid + 256 * i;                 // 1024 float4 slots: 128 rows x 8
      int r = s >> 3, c4 = s & 7;
      ga[i] = *(const float4*)(A + (size_t)(row0 + r) * HH + k0 + c4 * 4);
    }
#pragma unroll
    for (int w = 0; w < NW; ++w)
#pragma unroll
      for (int i = 0; i < 2; ++i) {
        int s = tid + 256 * i;               // 512 slots: 64 rows x 8
        int r = s >> 3, c4 = s & 7;
        gb[w][i] = *(const float4*)(Wp[w] + (size_t)(col0 + r) * HH + k0 + c4 * 4);
      }
  };

  auto split4 = [](float4 v, PK4& h, PK4& l) {
    float f[4] = {v.x, v.y, v.z, v.w};
#pragma unroll
    for (int j = 0; j < 4; ++j) {
      __bf16 hb = (__bf16)f[j];
      h.b[j] = hb;
      l.b[j] = (__bf16)(f[j] - (float)hb);
    }
  };

  auto store_lds = [&](int buf) {
#pragma unroll
    for (int i = 0; i < 4; ++i) {
      int s = tid + 256 * i;
      int r = s >> 3, c4 = s & 7;
      PK4 h, l;
      split4(ga[i], h, l);
      *(uint2*)&Ahi[buf][r][c4 * 4] = h.u;
      *(uint2*)&Alo[buf][r][c4 * 4] = l.u;
    }
#pragma unroll
    for (int w = 0; w < NW; ++w)
#pragma unroll
      for (int i = 0; i < 2; ++i) {
        int s = tid + 256 * i;
        int r = s >> 3, c4 = s & 7;
        PK4 h, l;
        split4(gb[w][i], h, l);
        *(uint2*)&Bhi[buf][w][r][c4 * 4] = h.u;
        *(uint2*)&Blo[buf][w][r][c4 * 4] = l.u;
      }
  };

  const int NK = HH / KC;  // 64
  load_global(0);
  store_lds(0);
  __syncthreads();

  for (int kc = 0; kc < NK; ++kc) {
    const int buf = kc & 1;
    if (kc + 1 < NK) load_global(kc + 1);

    bf16x16 ah[2], al[2];
#pragma unroll
    for (int mf = 0; mf < 2; ++mf) {
      const int r = waveM * 32 + mf * 16 + l16;
      ah[mf] = load_frag_a(&Ahi[buf][r][0], hf);
      al[mf] = load_frag_a(&Alo[buf][r][0], hf);
    }
#pragma unroll
    for (int w = 0; w < NW; ++w) {
#pragma unroll
      for (int nf = 0; nf < 2; ++nf) {
        const int n = waveN * 32 + nf * 16 + l16;
        bf16x16 bh = load_frag_b(&Bhi[buf][w][n][0], hf);
        bf16x16 bl = load_frag_b(&Blo[buf][w][n][0], hf);
#pragma unroll
        for (int mf = 0; mf < 2; ++mf) {
          acc[w][mf][nf] = wmma_bf16(ah[mf], bh, acc[w][mf][nf]);  // hi*hi
          acc[w][mf][nf] = wmma_bf16(al[mf], bh, acc[w][mf][nf]);  // lo*hi
          acc[w][mf][nf] = wmma_bf16(ah[mf], bl, acc[w][mf][nf]);  // hi*lo
        }
      }
    }
    if (kc + 1 < NK) store_lds((kc + 1) & 1);
    __syncthreads();
  }

  // Epilogue. C/D layout: VGPR r <-> m = r + 8*half (lanes 0-15 vs 16-31), n = lane&15.
#pragma unroll
  for (int mf = 0; mf < 2; ++mf) {
#pragma unroll
    for (int nf = 0; nf < 2; ++nf) {
      const int col = col0 + waveN * 32 + nf * 16 + l16;
#pragma unroll
      for (int r = 0; r < 8; ++r) {
        const int row = row0 + waveM * 32 + mf * 16 + 8 * hf + r;
        if constexpr (EPI) {
          const float vv = acc[0][mf][nf][r];
          const float kk = acc[1][mf][nf][r];
          const float rr = acc[2][mf][nf][r];
          const float rs = 1.0f / (1.0f + expf(-rr));
          const float kv = kk * vv;
          const float4 l4 = *(const float4*)(lw + (size_t)row * 4);
          const float ls = l4.x + l4.y + l4.z + l4.w;
          const int cl = col - col0;
          const float wsum = l4.x * dec_s[0][cl] + l4.y * dec_s[1][cl] +
                             l4.z * dec_s[2][cl] + l4.w * dec_s[3][cl] + kv * ls;
          out0[(size_t)row * HH + col] = rs * wsum;
          if ((row & (TT - 1)) == (TT - 1))
            kv_last[(size_t)(row >> 12) * HH + col] = kv;  // T = 4096 = 2^12
        } else {
          out0[(size_t)row * HH + col] = acc[0][mf][nf][r];
        }
      }
    }
  }
}

__launch_bounds__(256) __global__
void lw_kernel(const float* __restrict__ x, const float* __restrict__ Wl,
               const float* __restrict__ bl, float* __restrict__ lw) {
  const int row  = blockIdx.x * 8 + (threadIdx.x >> 5);
  const int lane = threadIdx.x & 31;
  const float* xr = x + (size_t)row * HH;
  float a0 = 0.f, a1 = 0.f, a2 = 0.f, a3 = 0.f;
  for (int k = lane; k < HH; k += 32) {
    const float xv = xr[k];
    a0 += xv * Wl[k];
    a1 += xv * Wl[HH + k];
    a2 += xv * Wl[2 * HH + k];
    a3 += xv * Wl[3 * HH + k];
  }
#pragma unroll
  for (int off = 16; off > 0; off >>= 1) {
    a0 += __shfl_xor(a0, off, 32);
    a1 += __shfl_xor(a1, off, 32);
    a2 += __shfl_xor(a2, off, 32);
    a3 += __shfl_xor(a3, off, 32);
  }
  if (lane == 0) {
    const float z0 = a0 + bl[0], z1 = a1 + bl[1], z2 = a2 + bl[2], z3 = a3 + bl[3];
    const float m  = fmaxf(fmaxf(z0, z1), fmaxf(z2, z3));
    const float e0 = expf(z0 - m), e1 = expf(z1 - m), e2 = expf(z2 - m), e3 = expf(z3 - m);
    const float inv = 1.0f / (e0 + e1 + e2 + e3);
    float4 o;
    o.x = e0 * inv; o.y = e1 * inv; o.z = e2 * inv; o.w = e3 * inv;
    *(float4*)(lw + (size_t)row * 4) = o;
  }
}

__launch_bounds__(256) __global__
void decay_kernel(const float* __restrict__ state, const float* __restrict__ tdm,
                  float* __restrict__ decayed) {
  const int i  = blockIdx.x * 256 + threadIdx.x;  // < B*D*H
  const int dh = i % (DDEP * HH);
  decayed[i] = state[i] * expf(-expf(tdm[dh]));
}

__launch_bounds__(256) __global__
void state_kernel(const float* __restrict__ decayed, const float* __restrict__ kv_last,
                  float* __restrict__ new_state) {
  const int i = blockIdx.x * 256 + threadIdx.x;  // < B*D*H
  const int b = i / (DDEP * HH);
  const int h = i % HH;
  new_state[i] = decayed[i] + kv_last[(size_t)b * HH + h];
}

extern "C" void kernel_launch(void* const* d_in, const int* in_sizes, int n_in,
                              void* d_out, int out_size, void* d_ws, size_t ws_size,
                              hipStream_t stream) {
  (void)in_sizes; (void)n_in; (void)out_size; (void)ws_size;
  const float* x   = (const float*)d_in[0];
  const float* st  = (const float*)d_in[1];
  const float* Wv  = (const float*)d_in[2];
  const float* Wk  = (const float*)d_in[3];
  const float* Wr  = (const float*)d_in[4];
  const float* Wo  = (const float*)d_in[5];
  const float* Wl  = (const float*)d_in[6];
  const float* bl  = (const float*)d_in[7];
  const float* tdm = (const float*)d_in[8];

  float* out       = (float*)d_out;                       // [B,T,H]
  float* new_state = out + (size_t)BBATCH * TT * HH;      // [B,D,H]

  // workspace: a_buf [M,H] f32 | lw [M,4] | decayed [B,D,H] | kv_last [B,H]
  float* a_buf   = (float*)d_ws;
  float* lwbuf   = a_buf + (size_t)BBATCH * TT * HH;
  float* decayed = lwbuf + (size_t)BBATCH * TT * DDEP;
  float* kv_last = decayed + (size_t)BBATCH * DDEP * HH;

  const int M = BBATCH * TT;  // 16384
  decay_kernel<<<(BBATCH * DDEP * HH) / 256, 256, 0, stream>>>(st, tdm, decayed);
  lw_kernel<<<M / 8, 256, 0, stream>>>(x, Wl, bl, lwbuf);

  dim3 grid(M / BM, HH / BN);
  gemm_kernel<3, true><<<grid, 256, 0, stream>>>(x, Wv, Wk, Wr, lwbuf, decayed,
                                                 a_buf, kv_last);
  state_kernel<<<(BBATCH * DDEP * HH) / 256, 256, 0, stream>>>(decayed, kv_last,
                                                               new_state);
  gemm_kernel<1, false><<<grid, 256, 0, stream>>>(a_buf, Wo, Wo, Wo, nullptr,
                                                  nullptr, out, nullptr);
}
